// DecoderLayer_34282428957085
// MI455X (gfx1250) — matrix operand order repaired and checked by reading, not compiled, not run on hardware
//
#include <hip/hip_runtime.h>

// ---------------------------------------------------------------------------
// CDNA5 (gfx1250, wave32) decoder layer: bf16 WMMA GEMMs + flash attention.
// ~240 GFLOP, compute-bound -> all matmuls on v_wmma_f32_16x16x32_bf16.
// f32->bf16 conversion hoisted into one-shot prep kernels so the GEMM hot
// loop stages pure bf16 copies; global loads software-pipelined past WMMAs.
// ---------------------------------------------------------------------------

typedef __attribute__((ext_vector_type(16))) __bf16 v16bf;
typedef __attribute__((ext_vector_type(8)))  float  v8f;

union FragU { v16bf v; unsigned u[8]; };
union Pk4 { __bf16 h[4]; uint2 u; };

__device__ inline v8f zero8() {
  v8f z;
#pragma unroll
  for (int i = 0; i < 8; ++i) z[i] = 0.f;
  return z;
}

// float -> bf16 round-to-nearest-even
__device__ inline __bf16 f2bf(float f) {
  union { float f; unsigned u; } a; a.f = f;
  unsigned r = a.u + 0x7fffu + ((a.u >> 16) & 1u);
  union { unsigned short s; __bf16 h; } o; o.s = (unsigned short)(r >> 16);
  return o.h;
}

// A-operand fragment (16x32 bf16, M x K), same K order as the B fragment:
// lane L: M = L%16, hi = L/16; VGPR v holds K = hi*16 + 2v (+1).
__device__ inline v16bf ld_frag_rowK(const __bf16* base, int ld, int row0, int k0) {
  const int lane = threadIdx.x & 31;
  const int hi = lane >> 4;
  const __bf16* p = base + (row0 + (lane & 15)) * ld + k0 + hi * 16;
  FragU f;
#pragma unroll
  for (int t = 0; t < 8; ++t) {
    f.u[t] = *(const unsigned*)(p + 2 * t);
  }
  return f.v;
}

// B-operand fragment (32x16 bf16, K x N) read from an N-major LDS tile
// (Bs[n][k], contiguous K). Same K order as A, not the ISA table order: lane L: N = L%16, hi = L/16;
// VGPR v holds K = hi*16 + 2v (+1).
__device__ inline v16bf ld_frag_colK(const __bf16* base, int ld, int n0, int k0) {
  const int lane = threadIdx.x & 31;
  const int hi = lane >> 4;
  const __bf16* p = base + (n0 + (lane & 15)) * ld + k0 + hi * 16;
  FragU f;
#pragma unroll
  for (int t = 0; t < 8; ++t) f.u[t] = *(const unsigned*)(p + 2 * t);
  return f.v;
}

__device__ inline float redmax16(float x) {
#pragma unroll
  for (int m = 8; m >= 1; m >>= 1) x = fmaxf(x, __shfl_xor(x, m, 32));
  return x;
}
__device__ inline float redsum16(float x) {
#pragma unroll
  for (int m = 8; m >= 1; m >>= 1) x += __shfl_xor(x, m, 32);
  return x;
}

// ---------------------------------------------------------------------------
// Prep kernels (one-shot, bandwidth-trivial):
//  cvt_bf16:      f32 activations -> bf16
//  wcvt_transpose: f32 weights (generalized [*,K,*] layout) -> bf16 Wt[N][K]
//   W element (k, n) at W[(n/n_div)*head_stride + k*k_stride + (n%n_div)]
// ---------------------------------------------------------------------------
__global__ __launch_bounds__(256) void cvt_bf16(const float* __restrict__ x,
                                                __bf16* __restrict__ y) {
  const long i = ((long)blockIdx.x * 256 + threadIdx.x) * 4;
  float4 d = *(const float4*)(x + i);
  Pk4 o;
  o.h[0] = f2bf(d.x); o.h[1] = f2bf(d.y);
  o.h[2] = f2bf(d.z); o.h[3] = f2bf(d.w);
  *(uint2*)(y + i) = o.u;
}

__global__ __launch_bounds__(256) void wcvt_transpose(
    const float* __restrict__ W, __bf16* __restrict__ Wt,
    int N, int K, int n_div, long head_stride, long k_stride) {
  const int gid = blockIdx.x * 256 + threadIdx.x;
  const int k = gid / (N / 4);
  const int n = (gid % (N / 4)) * 4;
  const float* p = W + (long)(n / n_div) * head_stride +
                   (long)k * k_stride + (n % n_div);
  float4 d = *(const float4*)p;
  Wt[(long)(n + 0) * K + k] = f2bf(d.x);
  Wt[(long)(n + 1) * K + k] = f2bf(d.y);
  Wt[(long)(n + 2) * K + k] = f2bf(d.z);
  Wt[(long)(n + 3) * K + k] = f2bf(d.w);
}

// ---------------------------------------------------------------------------
// GEMM: C[M,N] = act(A[M,K]bf16 * Wt[N][K]bf16 + bias)
// BM=BN=128, BK=64, 256 threads = 8 waves, each wave owns 32x64 of C.
// Next tile's global loads issued before the WMMA block (reg prefetch).
// ---------------------------------------------------------------------------
#define BM 128
#define BN 128
#define BK 64
#define PAD 4

template <bool OUTBF16, bool RELU>
__global__ __launch_bounds__(256) void gemm_wmma(
    const __bf16* __restrict__ A, const __bf16* __restrict__ Wt,
    const float* __restrict__ bias, void* __restrict__ Cp,
    int M, int N, int K) {
  __shared__ __bf16 As[BM][BK + PAD];
  __shared__ __bf16 Bs[BN][BK + PAD];  // N-major, contiguous K

  const int t = threadIdx.x;
  const int lane = t & 31;
  const int hi = lane >> 4;
  const int m0 = blockIdx.y * BM;
  const int n0 = blockIdx.x * BN;
  const int w = t >> 5;
  const int wm = (w & 3) * 32;   // 4 wave-rows
  const int wn = (w >> 2) * 64;  // 2 wave-cols
  const int rt = t >> 4;         // staging row within 16-row stripe
  const int ct = (t & 15) * 4;   // staging col (4 bf16 = 8B)

  v8f acc[2][4];
#pragma unroll
  for (int i = 0; i < 2; ++i)
#pragma unroll
    for (int j = 0; j < 4; ++j) acc[i][j] = zero8();

  uint2 ra[8], rb[8];
  // prefetch tile 0
#pragma unroll
  for (int i = 0; i < 8; ++i) {
    ra[i] = *(const uint2*)(A + (long)(m0 + i * 16 + rt) * K + ct);
    rb[i] = *(const uint2*)(Wt + (long)(n0 + i * 16 + rt) * K + ct);
  }

  const int nk = K / BK;
  for (int kt = 0; kt < nk; ++kt) {
    __syncthreads();  // previous compute done reading LDS
#pragma unroll
    for (int i = 0; i < 8; ++i) {
      *(uint2*)&As[i * 16 + rt][ct] = ra[i];
      *(uint2*)&Bs[i * 16 + rt][ct] = rb[i];
    }
    __syncthreads();
    if (kt + 1 < nk) {  // issue next tile's loads; latency hides behind WMMA
      const int k0 = (kt + 1) * BK;
#pragma unroll
      for (int i = 0; i < 8; ++i) {
        ra[i] = *(const uint2*)(A + (long)(m0 + i * 16 + rt) * K + k0 + ct);
        rb[i] = *(const uint2*)(Wt + (long)(n0 + i * 16 + rt) * K + k0 + ct);
      }
    }
#pragma unroll
    for (int ks = 0; ks < BK; ks += 32) {
      v16bf a0 = ld_frag_rowK(&As[0][0], BK + PAD, wm, ks);
      v16bf a1 = ld_frag_rowK(&As[0][0], BK + PAD, wm + 16, ks);
#pragma unroll
      for (int nt = 0; nt < 4; ++nt) {
        v16bf b = ld_frag_colK(&Bs[0][0], BK + PAD, wn + nt * 16, ks);
        acc[0][nt] = __builtin_amdgcn_wmma_f32_16x16x32_bf16(
            false, a0, false, b, (short)0, acc[0][nt], false, false);
        acc[1][nt] = __builtin_amdgcn_wmma_f32_16x16x32_bf16(
            false, a1, false, b, (short)0, acc[1][nt], false, false);
      }
    }
  }

  // epilogue: bias + activation, C-layout -> global
#pragma unroll
  for (int mt = 0; mt < 2; ++mt) {
#pragma unroll
    for (int nt = 0; nt < 4; ++nt) {
      const int col = n0 + wn + nt * 16 + (lane & 15);
      const float bc = bias[col];
#pragma unroll
      for (int r = 0; r < 8; ++r) {
        const int row = m0 + wm + mt * 16 + r + hi * 8;
        float v = acc[mt][nt][r] + bc;
        if constexpr (RELU) v = fmaxf(v, 0.f);
        if constexpr (OUTBF16)
          ((__bf16*)Cp)[(long)row * N + col] = f2bf(v);
        else
          ((float*)Cp)[(long)row * N + col] = v;
      }
    }
  }
}

// ---------------------------------------------------------------------------
// Flash attention: one block = 128 query rows of one (b, h).
// Q,K,V,O all bf16 in [B*S, H*64] layout (head h at column h*64).
// KV tiles register-prefetched one iteration ahead.
// ---------------------------------------------------------------------------
#define ADH 64

__global__ __launch_bounds__(256) void attn_wmma(
    const __bf16* __restrict__ Q, const __bf16* __restrict__ Kb,
    const __bf16* __restrict__ Vb, __bf16* __restrict__ O,
    int S, int H, float scale, int causal) {
  __shared__ __bf16 Qs[128][ADH + PAD];
  __shared__ __bf16 Ks[64][ADH + PAD];   // [key][dh] row-major
  __shared__ __bf16 Vt[ADH][64 + PAD];   // [dh][key] transposed
  __shared__ __bf16 Ps[128][64 + PAD];   // probabilities, per-wave rows

  const int t = threadIdx.x;
  const int w = t >> 5;
  const int lane = t & 31;
  const int hi = lane >> 4;
  const int b = blockIdx.z, h = blockIdx.y;
  const int q0 = blockIdx.x * 128;
  const long rs = (long)H * ADH;  // 1024
  const __bf16* Qg = Q + (long)b * S * rs + h * ADH;
  const __bf16* Kg = Kb + (long)b * S * rs + h * ADH;
  const __bf16* Vg = Vb + (long)b * S * rs + h * ADH;
  const int rt = t >> 4;         // 0..15 (key row stripe)
  const int ct = (t & 15) * 4;   // dh col

  // stage Q tile (128 x 64)
#pragma unroll
  for (int i = 0; i < 8; ++i) {
    const int row = i * 16 + rt;
    *(uint2*)&Qs[row][ct] = *(const uint2*)(Qg + (long)(q0 + row) * rs + ct);
  }

  v8f Oc[4];
  float m_s[8], l_s[8];
#pragma unroll
  for (int d = 0; d < 4; ++d) Oc[d] = zero8();
#pragma unroll
  for (int r = 0; r < 8; ++r) { m_s[r] = -3e38f; l_s[r] = 0.f; }

  const int kv_end = causal ? min(S, q0 + 128) : S;

  uint2 rk[4], rv[4];
#pragma unroll
  for (int i = 0; i < 4; ++i) {  // prefetch kv block 0
    const int row = i * 16 + rt;
    rk[i] = *(const uint2*)(Kg + (long)row * rs + ct);
    rv[i] = *(const uint2*)(Vg + (long)row * rs + ct);
  }

  for (int kv0 = 0; kv0 < kv_end; kv0 += 64) {
    __syncthreads();  // Q staged (iter 0) / prev PV done reading Ks,Vt
#pragma unroll
    for (int i = 0; i < 4; ++i) {
      const int row = i * 16 + rt;
      *(uint2*)&Ks[row][ct] = rk[i];
      const __bf16* e = (const __bf16*)&rv[i];
      Vt[ct + 0][row] = e[0];
      Vt[ct + 1][row] = e[1];
      Vt[ct + 2][row] = e[2];
      Vt[ct + 3][row] = e[3];
    }
    __syncthreads();
    if (kv0 + 64 < kv_end) {  // prefetch next kv block behind the math
#pragma unroll
      for (int i = 0; i < 4; ++i) {
        const int row = kv0 + 64 + i * 16 + rt;
        rk[i] = *(const uint2*)(Kg + (long)row * rs + ct);
        rv[i] = *(const uint2*)(Vg + (long)row * rs + ct);
      }
    }

    // S = Q * K^T (16 q-rows per wave x 64 keys; contract dh=64)
    v8f sc[4];
#pragma unroll
    for (int nt = 0; nt < 4; ++nt) sc[nt] = zero8();
#pragma unroll
    for (int ks = 0; ks < ADH; ks += 32) {
      v16bf a = ld_frag_rowK(&Qs[0][0], ADH + PAD, w * 16, ks);
#pragma unroll
      for (int nt = 0; nt < 4; ++nt) {
        v16bf bb = ld_frag_colK(&Ks[0][0], ADH + PAD, nt * 16, ks);
        sc[nt] = __builtin_amdgcn_wmma_f32_16x16x32_bf16(
            false, a, false, bb, (short)0, sc[nt], false, false);
      }
    }

    // online softmax (row stats replicated across each 16-lane half)
#pragma unroll
    for (int r = 0; r < 8; ++r) {
      const int qr = q0 + w * 16 + r + hi * 8;
      float mx = -3e38f;
#pragma unroll
      for (int nt = 0; nt < 4; ++nt) {
        float v = sc[nt][r] * scale;
        if (causal) {
          const int key = kv0 + nt * 16 + (lane & 15);
          if (key > qr) v = -1e9f;
        }
        sc[nt][r] = v;
        mx = fmaxf(mx, v);
      }
      mx = redmax16(mx);
      const float mnew = fmaxf(m_s[r], mx);
      const float alpha = __expf(m_s[r] - mnew);
      m_s[r] = mnew;
      float lsum = 0.f;
#pragma unroll
      for (int nt = 0; nt < 4; ++nt) {
        const float p = __expf(sc[nt][r] - mnew);
        sc[nt][r] = p;
        lsum += p;
      }
      l_s[r] = l_s[r] * alpha + redsum16(lsum);
#pragma unroll
      for (int d = 0; d < 4; ++d) Oc[d][r] = Oc[d][r] * alpha;
#pragma unroll
      for (int nt = 0; nt < 4; ++nt)
        Ps[w * 16 + r + hi * 8][nt * 16 + (lane & 15)] = f2bf(sc[nt][r]);
    }
    __syncthreads();

    // O += P * V (contract keys=64)
#pragma unroll
    for (int ks = 0; ks < 64; ks += 32) {
      v16bf a = ld_frag_rowK(&Ps[0][0], 64 + PAD, w * 16, ks);
#pragma unroll
      for (int d = 0; d < 4; ++d) {
        v16bf bb = ld_frag_colK(&Vt[0][0], 64 + PAD, d * 16, ks);
        Oc[d] = __builtin_amdgcn_wmma_f32_16x16x32_bf16(
            false, a, false, bb, (short)0, Oc[d], false, false);
      }
    }
  }

  // finalize: O /= l, write bf16 (concat-of-heads layout)
  __bf16* Og = O + (long)b * S * rs + h * ADH;
#pragma unroll
  for (int r = 0; r < 8; ++r) {
    const float inv = 1.f / l_s[r];
    const int row = q0 + w * 16 + r + hi * 8;
#pragma unroll
    for (int d = 0; d < 4; ++d)
      Og[(long)row * rs + d * 16 + (lane & 15)] = f2bf(Oc[d][r] * inv);
  }
}

// ---------------------------------------------------------------------------
// Host orchestration. Workspace layout (needs 132 MB):
//   [0,16M)   de_bf          (dead after q1 -> reused by y1)
//   [16M,32M) en_bf          (dead after k2/v2 -> reused by y1)
//   y1 = [0,32M) bf16 8192x2048 (FFN hidden, after de_bf/en_bf dead)
//   [32M,48M) qB  [48M,64M) kB  [64M,80M) vB
//   [80M,96M) h1  [96M,112M) h2
//   [112M,132M) bf16 transposed weights:
//     wq1 wk1 wv1 wq2 wk2 wv2 (2MB each), wf1 (4MB), wf2 (4MB)
// ---------------------------------------------------------------------------
extern "C" void kernel_launch(void* const* d_in, const int* in_sizes, int n_in,
                              void* d_out, int out_size, void* d_ws,
                              size_t ws_size, hipStream_t stream) {
  const float* de_x = (const float*)d_in[0];
  const float* en_x = (const float*)d_in[1];
  // d_in[2] = mask: causal triu(k=1); handled analytically in attn_wmma
  const float* a1w = (const float*)d_in[3];
  const float* a1b = (const float*)d_in[4];
  const float* a2w = (const float*)d_in[5];
  const float* a2b = (const float*)d_in[6];
  const float* w1 = (const float*)d_in[7];
  const float* b1 = (const float*)d_in[8];
  const float* w2 = (const float*)d_in[9];
  const float* b2 = (const float*)d_in[10];
  float* out = (float*)d_out;

  const int B = 8, S = 1024, D = 1024, H = 16, DH = 64, DFF = 2048;
  const int M = B * S;                  // 8192
  const long wsel = (long)H * D * DH;   // floats per q/k/v weight

  char* ws = (char*)d_ws;
  __bf16* de_bf = (__bf16*)(ws + ((size_t)0 << 20));
  __bf16* en_bf = (__bf16*)(ws + ((size_t)16 << 20));
  __bf16* y1    = (__bf16*)(ws + ((size_t)0 << 20));  // aliases de/en (dead)
  __bf16* qB = (__bf16*)(ws + ((size_t)32 << 20));
  __bf16* kB = (__bf16*)(ws + ((size_t)48 << 20));
  __bf16* vB = (__bf16*)(ws + ((size_t)64 << 20));
  __bf16* h1 = (__bf16*)(ws + ((size_t)80 << 20));
  __bf16* h2 = (__bf16*)(ws + ((size_t)96 << 20));
  __bf16* wq1 = (__bf16*)(ws + ((size_t)112 << 20));
  __bf16* wk1 = (__bf16*)(ws + ((size_t)114 << 20));
  __bf16* wv1 = (__bf16*)(ws + ((size_t)116 << 20));
  __bf16* wq2 = (__bf16*)(ws + ((size_t)118 << 20));
  __bf16* wk2 = (__bf16*)(ws + ((size_t)120 << 20));
  __bf16* wv2 = (__bf16*)(ws + ((size_t)122 << 20));
  __bf16* wf1 = (__bf16*)(ws + ((size_t)124 << 20));
  __bf16* wf2 = (__bf16*)(ws + ((size_t)128 << 20));

  const dim3 blk(256);
  const float scale = 0.125f;  // 1/sqrt(64)

  // ---- prep: activations f32->bf16; weights f32->bf16 [N][K] ----
  cvt_bf16<<<M * D / 1024, blk, 0, stream>>>(de_x, de_bf);
  cvt_bf16<<<M * D / 1024, blk, 0, stream>>>(en_x, en_bf);
  const int gw = (1024 / 4) * 1024 / 256;  // q/k/v weight: N=1024, K=1024
  wcvt_transpose<<<gw, blk, 0, stream>>>(a1w, wq1, 1024, 1024, DH, (long)D * DH, DH);
  wcvt_transpose<<<gw, blk, 0, stream>>>(a1w + wsel, wk1, 1024, 1024, DH, (long)D * DH, DH);
  wcvt_transpose<<<gw, blk, 0, stream>>>(a1w + 2 * wsel, wv1, 1024, 1024, DH, (long)D * DH, DH);
  wcvt_transpose<<<gw, blk, 0, stream>>>(a2w, wq2, 1024, 1024, DH, (long)D * DH, DH);
  wcvt_transpose<<<gw, blk, 0, stream>>>(a2w + wsel, wk2, 1024, 1024, DH, (long)D * DH, DH);
  wcvt_transpose<<<gw, blk, 0, stream>>>(a2w + 2 * wsel, wv2, 1024, 1024, DH, (long)D * DH, DH);
  wcvt_transpose<<<(DFF / 4) * 1024 / 256, blk, 0, stream>>>(
      w1, wf1, DFF, 1024, DFF, 0L, (long)DFF);
  wcvt_transpose<<<(1024 / 4) * DFF / 256, blk, 0, stream>>>(
      w2, wf2, 1024, DFF, 1024, 0L, (long)1024);

  const dim3 gP(1024 / BN, M / BM);   // (8, 64)
  const dim3 gF1(DFF / BN, M / BM);   // (16, 64)
  const dim3 gF2(1024 / BN, M / BM);  // (8, 64)
  const dim3 gA(S / 128, H, B);       // (8, 16, 8)

  // ---- block 1: Q from de_x (causal), K/V from en_x ----
  gemm_wmma<true, false><<<gP, blk, 0, stream>>>(de_bf, wq1, a1b, qB, M, 1024, D);
  gemm_wmma<true, false><<<gP, blk, 0, stream>>>(en_bf, wk1, a1b + H * DH, kB, M, 1024, D);
  gemm_wmma<true, false><<<gP, blk, 0, stream>>>(en_bf, wv1, a1b + 2 * H * DH, vB, M, 1024, D);
  attn_wmma<<<gA, blk, 0, stream>>>(qB, kB, vB, h1, S, H, scale, 1);

  // ---- block 2: Q from h1, K/V from en_x, no mask ----
  gemm_wmma<true, false><<<gP, blk, 0, stream>>>(h1, wq2, a2b, qB, M, 1024, D);
  gemm_wmma<true, false><<<gP, blk, 0, stream>>>(en_bf, wk2, a2b + H * DH, kB, M, 1024, D);
  gemm_wmma<true, false><<<gP, blk, 0, stream>>>(en_bf, wv2, a2b + 2 * H * DH, vB, M, 1024, D);
  attn_wmma<<<gA, blk, 0, stream>>>(qB, kB, vB, h2, S, H, scale, 0);

  // ---- FFN: relu(h2 @ w1 + b1) @ w2 + b2 ----  (en_bf/de_bf dead -> y1)
  gemm_wmma<true, true><<<gF1, blk, 0, stream>>>(h2, wf1, b1, y1, M, DFF, D);
  gemm_wmma<false, false><<<gF2, blk, 0, stream>>>(y1, wf2, b2, out, M, 1024, DFF);
}
